// View_51986284151376
// MI455X (gfx1250) — matrix operand order repaired and checked by reading, not compiled, not run on hardware
//
#include <hip/hip_runtime.h>
#include <math.h>

#define HH 320
#define WW 320
#define NS 32
#define RPB 8                 // rays per block = waves per block
#define PTS (RPB * NS)        // 256 points per block
#define HID 64

// d_ws layout (f16): [0,2048) W1 B-swizzled | [2048,6144) W2 | [6144,7168) W3
#define W1_OFF 0
#define W2_OFF 2048
#define W3_OFF 6144

typedef __attribute__((ext_vector_type(16))) _Float16 v16h;
typedef __attribute__((ext_vector_type(8)))  _Float16 v8h;
typedef __attribute__((ext_vector_type(8)))  float    v8f;

__device__ __forceinline__ v16h combine16(v8h lo, v8h hi) {
    v16h r;
#pragma unroll
    for (int i = 0; i < 8; ++i) { r[i] = lo[i]; r[i + 8] = hi[i]; }
    return r;
}

// ---- one-time weight swizzle into B-matrix layout: [block][half][nlane][i] ----
__global__ __launch_bounds__(256) void swizzle_weights_kernel(
    const float* __restrict__ W1g, const float* __restrict__ W2g,
    const float* __restrict__ W3g, _Float16* __restrict__ wsw)
{
    const int tid = threadIdx.x;
    // W1 (3x64): 1 K-chunk x 4 N-tiles, zero-padded K>=3
    for (int d = tid; d < 4 * 512; d += 256) {
        const int nt = d >> 9, hl = (d >> 8) & 1, n = (d >> 4) & 15, i = d & 15;
        const int k = hl * 16 + i;
        wsw[W1_OFF + d] = (k < 3) ? (_Float16)W1g[k * HID + nt * 16 + n] : (_Float16)0.0f;
    }
    // W2 (64x64): 2 K-chunks x 4 N-tiles (src-major: coalesced reads)
    for (int e = tid; e < HID * HID; e += 256) {
        const int k = e >> 6, n = e & 63;
        const int dst = (((k >> 5) * 4 + (n >> 4)) << 9) + (((k >> 4) & 1) << 8)
                      + ((n & 15) << 4) + (k & 15);
        wsw[W2_OFF + dst] = (_Float16)W2g[e];
    }
    // W3 (64x4): 2 K-chunks x 1 N-tile, zero-padded N>=4
    for (int d = tid; d < 2 * 512; d += 256) {
        const int c = d >> 9, hl = (d >> 8) & 1, n = (d >> 4) & 15, i = d & 15;
        const int k = c * 32 + hl * 16 + i;
        wsw[W3_OFF + d] = (n < 4) ? (_Float16)W3g[k * 4 + n] : (_Float16)0.0f;
    }
}

// ---- fully fused, barrier-free renderer: 1 ray per wave, lane = sample ----
__global__ __launch_bounds__(256) void nerf_fused_kernel(
    const float* __restrict__ Kc,  const float* __restrict__ Twc,
    const float* __restrict__ amn, const float* __restrict__ amx,
    const float* __restrict__ b1g, const float* __restrict__ b2g,
    const float* __restrict__ b3g, const _Float16* __restrict__ wsw,
    float* __restrict__ out)
{
    __shared__ __align__(16) _Float16 sH[PTS * HID];   // 32 KB, h1 -> h2 in place (wave-local)
    __shared__ __align__(16) float    sRaw[PTS * 4];   // 4 KB, activated (sigma,r,g,b)

    const int tid   = threadIdx.x;
    const int lane  = tid & 31;        // = sample index within the ray
    const int wv    = tid >> 5;        // = ray index within the block
    const int hlf   = lane >> 4;
    const int nlane = lane & 15;

    // ---- ray setup (all lanes of a wave share one ray) ----
    const int rayg = blockIdx.x * RPB + wv;
    const int py = rayg / WW, px = rayg % WW;

    const float fx = Kc[0], fy = Kc[4], cx = Kc[2], cy = Kc[5];
    float dxc = ((float)px + 0.5f - cx) / fx;
    float dyc = ((float)py + 0.5f - cy) / fy;
    const float il = 1.0f / sqrtf(dxc * dxc + dyc * dyc + 1.0f);
    dxc *= il; dyc *= il; const float dzc = il;

    const float dwx = Twc[0] * dxc + Twc[1] * dyc + Twc[2]  * dzc;
    const float dwy = Twc[4] * dxc + Twc[5] * dyc + Twc[6]  * dzc;
    const float dwz = Twc[8] * dxc + Twc[9] * dyc + Twc[10] * dzc;
    const float ox = Twc[3], oy = Twc[7], oz = Twc[11];

    const float ivx = 1.0f / dwx, ivy = 1.0f / dwy, ivz = 1.0f / dwz;
    const float t1x = (amn[0] - ox) * ivx, t2x = (amx[0] - ox) * ivx;
    const float t1y = (amn[1] - oy) * ivy, t2y = (amx[1] - oy) * ivy;
    const float t1z = (amn[2] - oz) * ivz, t2z = (amx[2] - oz) * ivz;
    const float tnear = fmaxf(fmaxf(fminf(t1x, t2x), fminf(t1y, t2y)), fminf(t1z, t2z));
    const float tfar  = fminf(fminf(fmaxf(t1x, t2x), fmaxf(t1y, t2y)), fmaxf(t1z, t2z));
    const int hit = tnear < tfar;
    const float tn = hit ? tnear : 0.0f;
    const float tf = hit ? tfar  : 1.0f;

    const float ts = tn + (tf - tn) * (((float)lane + 0.5f) / (float)NS);
    const float nx = ((ox + ts * dwx) - amn[0]) / (amx[0] - amn[0]) * 2.0f - 1.0f;
    const float ny = ((oy + ts * dwy) - amn[1]) / (amx[1] - amn[1]) * 2.0f - 1.0f;
    const float nz = ((oz + ts * dwz) - amn[2]) / (amx[2] - amn[2]) * 2.0f - 1.0f;

    // ---- B operands: coalesced global_load_b128 from pre-swizzled weights ----
    v16h B1v[4], B2v[2][4], B3v[2];
#pragma unroll
    for (int nt = 0; nt < 4; ++nt) {
        const _Float16* p = wsw + W1_OFF + (nt << 9) + (hlf << 8) + (nlane << 4);
        B1v[nt] = combine16(*(const v8h*)p, *(const v8h*)(p + 8));
    }
#pragma unroll
    for (int c = 0; c < 2; ++c)
#pragma unroll
        for (int nt = 0; nt < 4; ++nt) {
            const _Float16* p = wsw + W2_OFF + ((c * 4 + nt) << 9) + (hlf << 8) + (nlane << 4);
            B2v[c][nt] = combine16(*(const v8h*)p, *(const v8h*)(p + 8));
        }
#pragma unroll
    for (int c = 0; c < 2; ++c) {
        const _Float16* p = wsw + W3_OFF + (c << 9) + (hlf << 8) + (nlane << 4);
        B3v[c] = combine16(*(const v8h*)p, *(const v8h*)(p + 8));
    }
    float b1n[4], b2n[4];
#pragma unroll
    for (int nt = 0; nt < 4; ++nt) { b1n[nt] = b1g[16 * nt + nlane]; b2n[nt] = b2g[16 * nt + nlane]; }
    const float b3n = b3g[nlane & 3];

    v8f zacc;
#pragma unroll
    for (int r8 = 0; r8 < 8; ++r8) zacc[r8] = 0.0f;

    // ---- fused 3-layer MLP over the wave's 2 row-tiles (16 samples each) ----
#pragma unroll
    for (int tt = 0; tt < 2; ++tt) {
        const int T = wv * 2 + tt;

        // layer 1: A built in registers via shuffles (row m = lane tt*16+nlane's xyz)
        const int src = tt * 16 + nlane;
        const float sx = __shfl(nx, src), sy = __shfl(ny, src), sz = __shfl(nz, src);
        v16h a1;
#pragma unroll
        for (int i = 0; i < 16; ++i) a1[i] = (_Float16)0.0f;
        if (hlf == 0) {
            a1[0] = (_Float16)sx; a1[1] = (_Float16)sy; a1[2] = (_Float16)sz;
        }
        v8f acc[4];
#pragma unroll
        for (int nt = 0; nt < 4; ++nt)
            acc[nt] = __builtin_amdgcn_wmma_f32_16x16x32_f16(
                false, a1, false, B1v[nt], (short)0, zacc, false, false);
#pragma unroll
        for (int nt = 0; nt < 4; ++nt)
#pragma unroll
            for (int r8 = 0; r8 < 8; ++r8)
                sH[(T * 16 + r8 + 8 * hlf) * HID + 16 * nt + nlane] =
                    (_Float16)fmaxf(acc[nt][r8] + b1n[nt], 0.0f);

        // layer 2: K=64 (2 chunks), N=64 (4 tiles)
        {
            const _Float16* rowp = &sH[(T * 16 + nlane) * HID];
            const v16h a0  = combine16(*(const v8h*)(rowp +      16 * hlf),
                                       *(const v8h*)(rowp + 8 + 16 * hlf));
            const v16h a1c = combine16(*(const v8h*)(rowp + 32 + 16 * hlf),
                                       *(const v8h*)(rowp + 40 + 16 * hlf));
#pragma unroll
            for (int nt = 0; nt < 4; ++nt) {
                acc[nt] = __builtin_amdgcn_wmma_f32_16x16x32_f16(
                    false, a0, false, B2v[0][nt], (short)0, zacc, false, false);
                acc[nt] = __builtin_amdgcn_wmma_f32_16x16x32_f16(
                    false, a1c, false, B2v[1][nt], (short)0, acc[nt], false, false);
            }
#pragma unroll
            for (int nt = 0; nt < 4; ++nt)
#pragma unroll
                for (int r8 = 0; r8 < 8; ++r8)
                    sH[(T * 16 + r8 + 8 * hlf) * HID + 16 * nt + nlane] =
                        (_Float16)fmaxf(acc[nt][r8] + b2n[nt], 0.0f);
        }

        // layer 3: K=64 (2 chunks), N=4 padded to 16; in-register activations
        {
            const _Float16* rowp = &sH[(T * 16 + nlane) * HID];
            const v16h a0  = combine16(*(const v8h*)(rowp +      16 * hlf),
                                       *(const v8h*)(rowp + 8 + 16 * hlf));
            const v16h a1c = combine16(*(const v8h*)(rowp + 32 + 16 * hlf),
                                       *(const v8h*)(rowp + 40 + 16 * hlf));
            v8f acc3 = __builtin_amdgcn_wmma_f32_16x16x32_f16(
                false, a0, false, B3v[0], (short)0, zacc, false, false);
            acc3 = __builtin_amdgcn_wmma_f32_16x16x32_f16(
                false, a1c, false, B3v[1], (short)0, acc3, false, false);
            if (nlane < 4) {
#pragma unroll
                for (int r8 = 0; r8 < 8; ++r8) {
                    const float x = acc3[r8] + b3n;
                    const float y = (nlane == 0)
                        ? fmaxf(x, 0.0f) + log1pf(expf(-fabsf(x)))   // softplus -> sigma
                        : 1.0f / (1.0f + expf(-x));                  // sigmoid  -> rgb
                    sRaw[(T * 16 + r8 + 8 * hlf) * 4 + nlane] = y;
                }
            }
        }
    }

    // ---- wave-parallel compositing: prefix-sum + butterfly reduction ----
    {
        const float* rw = &sRaw[tid * 4];
        const float sg = rw[0], cr = rw[1], cg = rw[2], cb = rw[3];
        const float dt = (tf - tn) / (float)NS;
        const float delta = (lane == NS - 1) ? 0.5f * dt : dt;
        const float sa = sg * delta;

        float cum = sa;                       // inclusive prefix sum of sa over lanes
#pragma unroll
        for (int d = 1; d < 32; d <<= 1) {
            const float t = __shfl_up(cum, d);
            if (lane >= d) cum += t;
        }
        const float Tp = expf(-(cum - sa));   // transmittance before this sample
        const float w  = Tp * (1.0f - expf(-sa));
        float wr = w * cr, wg = w * cg, wb = w * cb;
#pragma unroll
        for (int d = 16; d >= 1; d >>= 1) {   // butterfly sum across the wave
            wr += __shfl_xor(wr, d);
            wg += __shfl_xor(wg, d);
            wb += __shfl_xor(wb, d);
        }
        const float total = __shfl(cum, 31);
        if (lane == 0) {
            float alpha = 1.0f - expf(-total);
            if (!hit) { wr = wg = wb = 0.0f; alpha = 0.0f; }
            out[0 * HH * WW + rayg] = wr;
            out[1 * HH * WW + rayg] = wg;
            out[2 * HH * WW + rayg] = wb;
            out[3 * HH * WW + rayg] = alpha;
        }
    }
}

extern "C" void kernel_launch(void* const* d_in, const int* in_sizes, int n_in,
                              void* d_out, int out_size, void* d_ws, size_t ws_size,
                              hipStream_t stream) {
    (void)in_sizes; (void)n_in; (void)out_size; (void)ws_size;
    const float* Kc  = (const float*)d_in[0];
    const float* Twc = (const float*)d_in[1];
    const float* amn = (const float*)d_in[2];
    const float* amx = (const float*)d_in[3];
    const float* W1  = (const float*)d_in[4];
    const float* b1  = (const float*)d_in[5];
    const float* W2  = (const float*)d_in[6];
    const float* b2  = (const float*)d_in[7];
    const float* W3  = (const float*)d_in[8];
    const float* b3  = (const float*)d_in[9];
    float* out = (float*)d_out;
    _Float16* wsw = (_Float16*)d_ws;          // needs 7168 * 2 = 14336 bytes

    swizzle_weights_kernel<<<dim3(1), dim3(256), 0, stream>>>(W1, W2, W3, wsw);
    nerf_fused_kernel<<<dim3((HH * WW) / RPB), dim3(256), 0, stream>>>(
        Kc, Twc, amn, amx, b1, b2, b3, wsw, out);
}
